// GBM_50233937494185
// MI455X (gfx1250) — matrix operand order repaired and checked by reading, not compiled, not run on hardware
//
#include <hip/hip_runtime.h>
#include <hip/hip_bf16.h>
#include <math.h>

typedef __attribute__((ext_vector_type(16))) _Float16 v16h;
typedef __attribute__((ext_vector_type(8)))  float    v8f;

#define DMODEL  256
#define TSTEPS  8
#define NREG    960
#define RFLAT   1024
#define ROWS    (TSTEPS * NREG)   // 7680
#define DINNER  512
#define DSTATE  16
#define DTRANK  16
#define NHEADS  8
#define DHEAD   32

__device__ __forceinline__ void cvt4(v16h& v, int o, float4 f)
{
    v[o + 0] = (_Float16)f.x;
    v[o + 1] = (_Float16)f.y;
    v[o + 2] = (_Float16)f.z;
    v[o + 3] = (_Float16)f.w;
}

// Load one 16x32 A fragment (this lane's 16 elements) -- fully unguarded,
// vectorized float4 loads. Ap already points at A + m*lda + grp*8.
__device__ __forceinline__ void load_a(const float* __restrict__ Ap, int kk, v16h& a)
{
    const float4* p0 = (const float4*)(Ap + kk);        // K = kk+grp*8 .. +7
    const float4* p1 = (const float4*)(Ap + kk + 16);   // K = kk+16+grp*8 .. +7
    cvt4(a, 0, p0[0]); cvt4(a, 4,  p0[1]);
    cvt4(a, 8, p1[0]); cvt4(a, 12, p1[1]);
}

// Load two 32x16 B fragments (cols n0,n1). BLAYOUT 0: B[n*ldb+k] (contig in k);
// BLAYOUT 1: B[k*ldb+n] (strided).
template <int BLAYOUT>
__device__ __forceinline__ void load_b(const float* __restrict__ B0,
                                       const float* __restrict__ B1,
                                       long ldb, int kk, int grp, v16h& b0, v16h& b1)
{
    const int kb = kk + grp * 8;
    if (BLAYOUT == 0) {
        const float4* p0 = (const float4*)(B0 + kb);
        const float4* p1 = (const float4*)(B1 + kb);
        cvt4(b0, 0, p0[0]); cvt4(b0, 4, p0[1]); cvt4(b0, 8, p0[4]); cvt4(b0, 12, p0[5]);
        cvt4(b1, 0, p1[0]); cvt4(b1, 4, p1[1]); cvt4(b1, 8, p1[4]); cvt4(b1, 12, p1[5]);
    } else {
        #pragma unroll
        for (int e = 0; e < 16; ++e) {
            b0[e] = (_Float16)B0[(long)(kb + e + (e >= 8 ? 8 : 0)) * ldb];
            b1[e] = (_Float16)B1[(long)(kb + e + (e >= 8 ? 8 : 0)) * ldb];
        }
    }
}

// ---------------------------------------------------------------------------
// Batched GEMM via WMMA f16->f32. One wave computes a 16(M) x 32(N) tile.
// Requirements: K % 32 == 0, M % 16 == 0, base pointers & leading dims
// 16-byte aligned. N handled by clamping the load column (epilogue masks).
//   C[m,n] = act( scale * sum_k A[m,k]*Bmat[k,n] + bias[n] )
//   ACT: 0 none, 1 relu, 2 softplus
// k-loop is unrolled 2x with independent SSA register groups so the second
// group's loads overlap the first group's WMMAs without register copies.
// ---------------------------------------------------------------------------
template <int BLAYOUT, int ACT>
__global__ __launch_bounds__(32) void gemm_wmma_kernel(
    const float* __restrict__ A, long lda, long strideA,
    const float* __restrict__ B, long ldb, long strideB,
    float* __restrict__ C, long ldc, long strideC,
    int M, int N, int K,
    const float* __restrict__ bias, float scale)
{
    const int mt = blockIdx.x, nt = blockIdx.y, bz = blockIdx.z;
    A += (long)bz * strideA;
    B += (long)bz * strideB;
    C += (long)bz * strideC;

    const int lane = threadIdx.x;
    const int l16  = lane & 15;
    const int grp  = lane >> 4;                 // 0 or 1
    const int m  = mt * 16 + l16;
    const int n0 = nt * 32 + l16;
    const int n1 = n0 + 16;
    const int n0c = (n0 < N) ? n0 : (N - 1);    // clamp: loads stay in bounds
    const int n1c = (n1 < N) ? n1 : (N - 1);

    const float* Ap = A + (long)m * lda + grp * 8;
    const float* B0 = (BLAYOUT == 0) ? (B + (long)n0c * ldb) : (B + n0c);
    const float* B1 = (BLAYOUT == 0) ? (B + (long)n1c * ldb) : (B + n1c);

    v8f acc0 = {}, acc1 = {};
    int kk = 0;
    for (; kk + 64 <= K; kk += 64) {
        v16h a0, p0, p1, a1, q0, q1;
        load_a(Ap, kk, a0);
        load_b<BLAYOUT>(B0, B1, ldb, kk, grp, p0, p1);
        load_a(Ap, kk + 32, a1);
        load_b<BLAYOUT>(B0, B1, ldb, kk + 32, grp, q0, q1);
        acc0 = __builtin_amdgcn_wmma_f32_16x16x32_f16(
                   false, a0, false, p0, (short)0, acc0, false, false);
        acc1 = __builtin_amdgcn_wmma_f32_16x16x32_f16(
                   false, a0, false, p1, (short)0, acc1, false, false);
        acc0 = __builtin_amdgcn_wmma_f32_16x16x32_f16(
                   false, a1, false, q0, (short)0, acc0, false, false);
        acc1 = __builtin_amdgcn_wmma_f32_16x16x32_f16(
                   false, a1, false, q1, (short)0, acc1, false, false);
    }
    if (kk < K) {                               // K % 64 == 32 tail (wave-uniform)
        v16h a0, p0, p1;
        load_a(Ap, kk, a0);
        load_b<BLAYOUT>(B0, B1, ldb, kk, grp, p0, p1);
        acc0 = __builtin_amdgcn_wmma_f32_16x16x32_f16(
                   false, a0, false, p0, (short)0, acc0, false, false);
        acc1 = __builtin_amdgcn_wmma_f32_16x16x32_f16(
                   false, a0, false, p1, (short)0, acc1, false, false);
    }

    const float bb0 = (bias && n0 < N) ? bias[n0] : 0.0f;
    const float bb1 = (bias && n1 < N) ? bias[n1] : 0.0f;
    #pragma unroll
    for (int r = 0; r < 8; ++r) {
        const int mr = mt * 16 + grp * 8 + r;   // D layout: M = 8*(lane/16)+vgpr
        float v0 = acc0[r] * scale + bb0;
        float v1 = acc1[r] * scale + bb1;
        if (ACT == 1) { v0 = fmaxf(v0, 0.0f); v1 = fmaxf(v1, 0.0f); }
        if (ACT == 2) {
            v0 = (v0 > 20.0f) ? v0 : log1pf(__expf(v0));
            v1 = (v1 > 20.0f) ? v1 : log1pf(__expf(v1));
        }
        if (mr < M) {
            if (n0 < N) C[(long)mr * ldc + n0] = v0;
            if (n1 < N) C[(long)mr * ldc + n1] = v1;
        }
    }
}

// ---------------------------------------------------------------------------
// zero-pad copy: dst[r, 0:scols] = src[r*sld + 0:scols], dst[r, scols:dcols] = 0
// ---------------------------------------------------------------------------
__global__ void pad_copy_kernel(const float* __restrict__ src, long sld,
                                float* __restrict__ dst, long rows, int scols, int dcols)
{
    long i = (long)blockIdx.x * blockDim.x + threadIdx.x;
    if (i >= rows * dcols) return;
    int c = (int)(i % dcols);
    long r = i / dcols;
    dst[i] = (c < scols) ? src[r * sld + c] : 0.0f;
}

// x (1,8,256,128,30) -> xr (8, 960, 1024) region tokens
__global__ void gather_regions_kernel(const float* __restrict__ x, float* __restrict__ xr)
{
    long i = (long)blockIdx.x * blockDim.x + threadIdx.x;
    if (i >= (long)ROWS * RFLAT) return;
    int f = (int)(i % RFLAT);
    long tr = i / RFLAT;
    int r = (int)(tr % NREG);
    int t = (int)(tr / NREG);
    int i2 = r % 15, i1 = (r / 15) & 7, i0 = r / 120;
    int c = f & 1, bq = (f >> 1) & 15, aq = f >> 5;
    int X = i0 * 32 + aq, Y = i1 * 16 + bq, Z = i2 * 2 + c;
    xr[i] = x[((long)t * 256 + X) * (128L * 30) + (long)Y * 30 + Z];
}

// logits (8, 960, 1024) -> out (1,8,256,128,30)
__global__ void scatter_regions_kernel(const float* __restrict__ logits, float* __restrict__ out)
{
    long i = (long)blockIdx.x * blockDim.x + threadIdx.x;
    if (i >= (long)TSTEPS * 256 * 128 * 30) return;
    int Z = (int)(i % 30);
    int Y = (int)((i / 30) % 128);
    int X = (int)((i / (30L * 128)) % 256);
    int t = (int)(i / (30L * 128 * 256));
    int i0 = X >> 5, aq = X & 31;
    int i1 = Y >> 4, bq = Y & 15;
    int i2 = Z >> 1, c = Z & 1;
    int r = (i0 * 8 + i1) * 15 + i2;
    int f = (aq * 16 + bq) * 2 + c;
    out[i] = logits[((long)t * NREG + r) * RFLAT + f];
}

// x <- rmsnorm(x + h) * w   (h may be null); one block (256 thr) per row
__global__ void add_rmsnorm_kernel(float* __restrict__ x, const float* __restrict__ h,
                                   const float* __restrict__ w)
{
    __shared__ float red[DMODEL];
    const long r = blockIdx.x;
    const int t = threadIdx.x;
    float v = x[r * DMODEL + t] + (h ? h[r * DMODEL + t] : 0.0f);
    red[t] = v * v;
    __syncthreads();
    for (int s = DMODEL / 2; s > 0; s >>= 1) {
        if (t < s) red[t] += red[t + s];
        __syncthreads();
    }
    float inv = rsqrtf(red[0] / (float)DMODEL + 1e-6f);
    x[r * DMODEL + t] = v * inv * w[t];
}

// in-place row softmax; one block (256 thr) per row of `cols`
__global__ void softmax_kernel(float* __restrict__ S, int cols)
{
    __shared__ float red[256];
    const int t = threadIdx.x;
    float* p = S + (long)blockIdx.x * cols;
    float mx = -1e30f;
    for (int c = t; c < cols; c += 256) mx = fmaxf(mx, p[c]);
    red[t] = mx;
    __syncthreads();
    for (int s = 128; s > 0; s >>= 1) { if (t < s) red[t] = fmaxf(red[t], red[t + s]); __syncthreads(); }
    mx = red[0];
    __syncthreads();
    float sum = 0.0f;
    for (int c = t; c < cols; c += 256) { float e = __expf(p[c] - mx); p[c] = e; sum += e; }
    red[t] = sum;
    __syncthreads();
    for (int s = 128; s > 0; s >>= 1) { if (t < s) red[t] += red[t + s]; __syncthreads(); }
    float inv = 1.0f / red[0];
    for (int c = t; c < cols; c += 256) p[c] *= inv;
}

// depthwise causal conv over time (D_CONV=4) + bias + silu
// xz: [7680, 1024] (xi = cols 0..511), rows ordered r = t*960 + n
__global__ void conv_silu_kernel(const float* __restrict__ xz, const float* __restrict__ cw,
                                 const float* __restrict__ cb, float* __restrict__ xs)
{
    long i = (long)blockIdx.x * blockDim.x + threadIdx.x;
    if (i >= (long)ROWS * DINNER) return;
    int c = (int)(i % DINNER);
    long r = i / DINNER;
    int n = (int)(r % NREG);
    int t = (int)(r / NREG);
    float acc = cb[c];
    #pragma unroll
    for (int j = 0; j < 4; ++j) {
        int ts = t - 3 + j;
        if (ts >= 0) acc += cw[c * 4 + j] * xz[((long)ts * NREG + n) * 1024 + c];
    }
    xs[r * DINNER + c] = acc / (1.0f + __expf(-acc));   // silu
}

// selective scan over T=8, one thread per (sequence n, channel c)
__global__ void mamba_scan_kernel(const float* __restrict__ xz,   // [7680,1024] z = cols 512..
                                  const float* __restrict__ xs,   // [7680,512]
                                  const float* __restrict__ dbl,  // [7680,48]  (dt|B|C)
                                  const float* __restrict__ dt,   // [7680,512] softplus'd
                                  const float* __restrict__ A_log,// [512,16]
                                  const float* __restrict__ Dp,   // [512]
                                  float* __restrict__ y)          // [7680,512]
{
    long i = (long)blockIdx.x * blockDim.x + threadIdx.x;
    if (i >= (long)NREG * DINNER) return;
    int c = (int)(i % DINNER);
    int n = (int)(i / DINNER);
    float Ac[DSTATE];
    #pragma unroll
    for (int s = 0; s < DSTATE; ++s) Ac[s] = -__expf(A_log[c * DSTATE + s]);
    float h[DSTATE] = {};
    const float Dc = Dp[c];
    for (int t = 0; t < TSTEPS; ++t) {
        long r = (long)t * NREG + n;
        float dtv = dt[r * DINNER + c];
        float xv  = xs[r * DINNER + c];
        float du  = dtv * xv;
        float yv = 0.0f;
        #pragma unroll
        for (int s = 0; s < DSTATE; ++s) {
            float dA = __expf(dtv * Ac[s]);
            float Bv = dbl[r * 48 + 16 + s];
            float Cv = dbl[r * 48 + 32 + s];
            h[s] = dA * h[s] + du * Bv;
            yv += h[s] * Cv;
        }
        yv += xv * Dc;
        float zv = xz[r * 1024 + 512 + c];
        yv *= zv / (1.0f + __expf(-zv));
        y[r * DINNER + c] = yv;
    }
}

// ---------------------------------------------------------------------------
// host side
// ---------------------------------------------------------------------------
static void launch_gemm(int blayout, int act,
                        const float* A, long lda, long sA,
                        const float* B, long ldb, long sB,
                        float* C, long ldc, long sC,
                        int M, int N, int K,
                        const float* bias, float scale, int batch, hipStream_t stream)
{
    dim3 grid((M + 15) / 16, (N + 31) / 32, batch), block(32);
#define GL(BL, AC) gemm_wmma_kernel<BL, AC><<<grid, block, 0, stream>>>( \
        A, lda, sA, B, ldb, sB, C, ldc, sC, M, N, K, bias, scale)
    if      (blayout == 0 && act == 0) GL(0, 0);
    else if (blayout == 0 && act == 1) GL(0, 1);
    else if (blayout == 0 && act == 2) GL(0, 2);
    else                               GL(1, 0);
#undef GL
}

extern "C" void kernel_launch(void* const* d_in, const int* in_sizes, int n_in,
                              void* d_out, int out_size, void* d_ws, size_t ws_size,
                              hipStream_t stream)
{
    const float* x          = (const float*)d_in[0];
    const float* enc_w      = (const float*)d_in[1];   // [256,1024]
    const float* enc_b      = (const float*)d_in[2];
    const float* enc_norm_w = (const float*)d_in[3];
    const float* dec_w      = (const float*)d_in[4];   // [1024,256]
    const float* dec_b      = (const float*)d_in[5];

    // workspace layout (floats)
    float* T1   = (float*)d_ws;                    // [7680,1024] xr / qkv / xz / mlp_h / logits
    float* TOK  = T1   + (long)ROWS * 1024;        // [7680,256]
    float* T2   = TOK  + (long)ROWS * DMODEL;      // [7680,512]
    float* S    = T2   + (long)ROWS * DINNER;      // [8,960,960] per-bt scores
    float* DBL  = S    + (long)NHEADS * NREG * NREG; // [7680,48]
    float* DT   = DBL  + (long)ROWS * 48;          // [7680,512]
    float* Y    = DT   + (long)ROWS * DINNER;      // [7680,512]
    float* PADA = Y    + (long)ROWS * DINNER;      // [7680,32]  dt zero-padded to K=32
    float* PADW = PADA + (long)ROWS * 32;          // [512,32]   dt_proj_w zero-padded

    const int EB = 256;
    const long tot_tok = (long)ROWS * RFLAT;

    // ---- encoder ----
    gather_regions_kernel<<<(tot_tok + EB - 1) / EB, EB, 0, stream>>>(x, T1);
    launch_gemm(0, 0, T1, RFLAT, 0, enc_w, RFLAT, 0, TOK, DMODEL, 0,
                ROWS, DMODEL, RFLAT, enc_b, 1.0f, 1, stream);
    add_rmsnorm_kernel<<<ROWS, DMODEL, 0, stream>>>(TOK, nullptr, enc_norm_w);

    // ---- layers ----
    for (int l = 0; l < 2; ++l) {
        const float* const* P = (const float* const*)(d_in + 6 + 20 * l);
        const float* attn_in_w  = P[0];  const float* attn_in_b  = P[1];
        const float* attn_out_w = P[2];  const float* attn_out_b = P[3];
        const float* norm1 = P[4]; const float* norm2 = P[5]; const float* norm3 = P[6];
        const float* mlp_w1 = P[7]; const float* mlp_b1 = P[8];
        const float* mlp_w2 = P[9]; const float* mlp_b2 = P[10];
        const float* in_proj_w = P[11];
        const float* conv_w = P[12]; const float* conv_b = P[13];
        const float* x_proj_w = P[14];
        const float* dt_proj_w = P[15]; const float* dt_proj_b = P[16];
        const float* A_log = P[17]; const float* Dparam = P[18];
        const float* out_proj_w = P[19];

        // --- MHA ---
        launch_gemm(0, 0, TOK, DMODEL, 0, attn_in_w, DMODEL, 0, T1, 3 * DMODEL, 0,
                    ROWS, 3 * DMODEL, DMODEL, attn_in_b, 1.0f, 1, stream); // qkv
        const float iscale = 0.17677669529663687f;  // 1/sqrt(32)
        for (int bt = 0; bt < TSTEPS; ++bt) {
            const float* qb = T1 + (long)bt * NREG * 3 * DMODEL;
            // S = (q @ k^T) / sqrt(dh), batched over heads via stride 32
            launch_gemm(0, 0, qb, 3 * DMODEL, DHEAD,
                        qb + DMODEL, 3 * DMODEL, DHEAD,
                        S, NREG, (long)NREG * NREG,
                        NREG, NREG, DHEAD, nullptr, iscale, NHEADS, stream);
            softmax_kernel<<<NHEADS * NREG, 256, 0, stream>>>(S, NREG);
            // O = softmax(S) @ V  (V is [K,N] row-major inside qkv)
            launch_gemm(1, 0, S, NREG, (long)NREG * NREG,
                        qb + 2 * DMODEL, 3 * DMODEL, DHEAD,
                        T2 + (long)bt * NREG * DMODEL, DMODEL, DHEAD,
                        NREG, DHEAD, NREG, nullptr, 1.0f, NHEADS, stream);
        }
        launch_gemm(0, 0, T2, DMODEL, 0, attn_out_w, DMODEL, 0, T1, DMODEL, 0,
                    ROWS, DMODEL, DMODEL, attn_out_b, 1.0f, 1, stream);
        add_rmsnorm_kernel<<<ROWS, DMODEL, 0, stream>>>(TOK, T1, norm1);

        // --- Mamba ---
        launch_gemm(0, 0, TOK, DMODEL, 0, in_proj_w, DMODEL, 0, T1, 2 * DINNER, 0,
                    ROWS, 2 * DINNER, DMODEL, nullptr, 1.0f, 1, stream); // xz
        conv_silu_kernel<<<((long)ROWS * DINNER + EB - 1) / EB, EB, 0, stream>>>(
            T1, conv_w, conv_b, T2);                                     // xs
        launch_gemm(0, 0, T2, DINNER, 0, x_proj_w, DINNER, 0, DBL, 48, 0,
                    ROWS, 48, DINNER, nullptr, 1.0f, 1, stream);         // dt|B|C
        // zero-pad dt input (cols 0..15 of DBL) and dt_proj_w to K=32
        pad_copy_kernel<<<((long)ROWS * 32 + EB - 1) / EB, EB, 0, stream>>>(
            DBL, 48, PADA, ROWS, DTRANK, 32);
        pad_copy_kernel<<<((long)DINNER * 32 + EB - 1) / EB, EB, 0, stream>>>(
            dt_proj_w, DTRANK, PADW, DINNER, DTRANK, 32);
        launch_gemm(0, 2, PADA, 32, 0, PADW, 32, 0, DT, DINNER, 0,
                    ROWS, DINNER, 32, dt_proj_b, 1.0f, 1, stream);       // softplus(dt)
        mamba_scan_kernel<<<((long)NREG * DINNER + EB - 1) / EB, EB, 0, stream>>>(
            T1, T2, DBL, DT, A_log, Dparam, Y);
        launch_gemm(0, 0, Y, DINNER, 0, out_proj_w, DINNER, 0, T2, DMODEL, 0,
                    ROWS, DMODEL, DINNER, nullptr, 1.0f, 1, stream);
        add_rmsnorm_kernel<<<ROWS, DMODEL, 0, stream>>>(TOK, T2, norm2);

        // --- MLP ---
        launch_gemm(0, 1, TOK, DMODEL, 0, mlp_w1, DMODEL, 0, T1, 2 * DMODEL, 0,
                    ROWS, 2 * DMODEL, DMODEL, mlp_b1, 1.0f, 1, stream);  // relu
        launch_gemm(0, 0, T1, 2 * DMODEL, 0, mlp_w2, 2 * DMODEL, 0, T2, DMODEL, 0,
                    ROWS, DMODEL, 2 * DMODEL, mlp_b2, 1.0f, 1, stream);
        add_rmsnorm_kernel<<<ROWS, DMODEL, 0, stream>>>(TOK, T2, norm3);
    }

    // ---- decoder ----
    launch_gemm(0, 0, TOK, DMODEL, 0, dec_w, DMODEL, 0, T1, RFLAT, 0,
                ROWS, RFLAT, DMODEL, dec_b, 1.0f, 1, stream);
    scatter_regions_kernel<<<(tot_tok + EB - 1) / EB, EB, 0, stream>>>(T1, (float*)d_out);
}
